// nonlocal_attention_37340445672047
// MI455X (gfx1250) — matrix operand order repaired and checked by reading, not compiled, not run on hardware
//
#include <hip/hip_runtime.h>
#include <math.h>

typedef __attribute__((ext_vector_type(16))) __bf16 v16bf;
typedef __attribute__((ext_vector_type(8)))  __bf16 v8bf;
typedef __attribute__((ext_vector_type(8)))  float  v8f;

#define T_FRAMES 7
#define C_CH 128
#define ICH 64
#define N_SP 4096
#define MB_K 256

static __device__ __forceinline__ v8f wmma_bf16(v16bf a, v16bf b, v8f c) {
  return __builtin_amdgcn_wmma_f32_16x16x32_bf16(false, a, false, b, (short)0, c, false, false);
}

// ---------------- GroupNorm: per (t, group) mean / rsqrt(var) ----------------
__global__ __launch_bounds__(256) void gn_stats_kernel(const float* __restrict__ x,
                                                       float* __restrict__ mu,
                                                       float* __restrict__ rs) {
  int t = blockIdx.x, g = blockIdx.y;          // 7 x 32
  const float* p = x + ((size_t)t * C_CH + g * 4) * N_SP;
  float s = 0.f, s2 = 0.f;
  for (int i = threadIdx.x; i < 4 * N_SP; i += 256) { float v = p[i]; s += v; s2 += v * v; }
  __shared__ float r1[256], r2[256];
  int tid = threadIdx.x;
  r1[tid] = s; r2[tid] = s2; __syncthreads();
  for (int st = 128; st > 0; st >>= 1) {
    if (tid < st) { r1[tid] += r1[tid + st]; r2[tid] += r2[tid + st]; }
    __syncthreads();
  }
  if (tid == 0) {
    float m = r1[0] * (1.f / 16384.f);
    float var = r2[0] * (1.f / 16384.f) - m * m;
    mu[t * 32 + g] = m;
    rs[t * 32 + g] = rsqrtf(var + 1e-6f);
  }
}

// hn transposed to [t][n][c] bf16; two adjacent channels per thread -> packed b32 stores
__global__ __launch_bounds__(256) void gn_apply_kernel(const float* __restrict__ x,
                                                       const float* __restrict__ mu,
                                                       const float* __restrict__ rs,
                                                       const float* __restrict__ gamma,
                                                       const float* __restrict__ beta,
                                                       __bf16* __restrict__ hnT) {
  int t = blockIdx.x, c0 = blockIdx.y * 2;     // grid (7, 64)
  int g = c0 >> 2;                             // c0, c0+1 are in the same group of 4
  float m = mu[t * 32 + g], r = rs[t * 32 + g];
  float a0 = gamma[c0] * r,     b0 = beta[c0] - m * a0;
  float a1 = gamma[c0 + 1] * r, b1 = beta[c0 + 1] - m * a1;
  const float* p0 = x + ((size_t)t * C_CH + c0) * N_SP;
  const float* p1 = p0 + N_SP;
  __bf16* o = hnT + (size_t)t * N_SP * C_CH + c0;
  for (int n = threadIdx.x; n < N_SP; n += 256) {
    union { __bf16 h[2]; unsigned u; } pk;
    pk.h[0] = (__bf16)(p0[n] * a0 + b0);
    pk.h[1] = (__bf16)(p1[n] * a1 + b1);
    *(unsigned*)(o + (size_t)n * C_CH) = pk.u;   // 4B aligned (c0 even)
  }
}

// ---------------- 1x1-conv projections via WMMA ----------------
// out[co][n] = sum_cin W[co][cin] * hn[t][cin][n] + bias[co]
// grid: (256 n-tiles, 15 combos: 0-6 theta, 7-13 g, 14 phi)   block: 128 (4 waves, co-tiles)
__global__ __launch_bounds__(128) void proj_kernel(const __bf16* __restrict__ hnT,
                                                   const float* __restrict__ theta_w, const float* __restrict__ theta_b,
                                                   const float* __restrict__ g_w,     const float* __restrict__ g_b,
                                                   const float* __restrict__ phi_w,   const float* __restrict__ phi_b,
                                                   __bf16* __restrict__ thetaT,       __bf16* __restrict__ gmat,
                                                   float*  __restrict__ phi_f,        __bf16* __restrict__ phi_h) {
  int j = blockIdx.y;
  const float *W, *Bb; int t, mode;
  if (j < 7)       { W = theta_w; Bb = theta_b; t = j;     mode = 0; }
  else if (j < 14) { W = g_w;     Bb = g_b;     t = j - 7; mode = 1; }
  else             { W = phi_w;   Bb = phi_b;   t = 3;     mode = 2; }

  int lane = threadIdx.x & 31, wav = threadIdx.x >> 5;
  bool lo = lane < 16;
  int lrow = lane & 15;
  int kb = lo ? 0 : 16;
  int co = wav * 16 + lrow;
  int n  = blockIdx.x * 16 + lrow;

  const __bf16* hrow = hnT + ((size_t)t * N_SP + n) * C_CH + (lo ? 0 : 16);
  const float*  wrow = W + (size_t)co * C_CH + kb;

  v8f acc = {0.f, 0.f, 0.f, 0.f, 0.f, 0.f, 0.f, 0.f};
#pragma unroll
  for (int kc = 0; kc < C_CH; kc += 32) {
    v16bf a;
#pragma unroll
    for (int i = 0; i < 8; i++) {
      a[i]     = (__bf16)wrow[kc + i];        // K = kc + kb + i
      a[i + 8] = (__bf16)wrow[kc + 8 + i];    // K = kc + kb + 8 + i
    }
    v16bf b = *(const v16bf*)(hrow + kc);     // K = kc + (lo?0:16) + i
    acc = wmma_bf16(a, b, acc);
  }

  int cbase = wav * 16 + (lo ? 0 : 8);        // D elem v -> co = cbase + v, n fixed per lane
#pragma unroll
  for (int v = 0; v < 8; v++) acc[v] += Bb[cbase + v];

  if (mode == 0) {                            // thetaT[t][m=n][co] bf16
    v8bf ob;
#pragma unroll
    for (int v = 0; v < 8; v++) ob[v] = (__bf16)acc[v];
    *(v8bf*)(thetaT + ((size_t)t * N_SP + n) * ICH + cbase) = ob;
  } else if (mode == 1) {                     // gmat[t][co][m=n] bf16
#pragma unroll
    for (int v = 0; v < 8; v++)
      gmat[((size_t)t * ICH + cbase + v) * N_SP + n] = (__bf16)acc[v];
  } else {                                    // phi[n][co] fp32 + bf16
    v8bf ob;
#pragma unroll
    for (int v = 0; v < 8; v++) { phi_f[(size_t)n * ICH + cbase + v] = acc[v]; ob[v] = (__bf16)acc[v]; }
    *(v8bf*)(phi_h + (size_t)n * ICH + cbase) = ob;
  }
}

// ---------------- fused non-local attention (flash-style, masked numerator) ----------------
// grid (64, 7), block 128 (4 waves x 16 rows). y[n][c] = sum_m softmax_m(f)[m]*w(n,m)*g[c][m]
__global__ __launch_bounds__(128) void attn_kernel(const __bf16* __restrict__ phi_h,
                                                   const __bf16* __restrict__ thetaT,
                                                   const __bf16* __restrict__ gmat,
                                                   const float* __restrict__ stdp,
                                                   float* __restrict__ corr) {
  __shared__ __align__(32) __bf16 plds[4][16][40];  // per-wave P tile, padded rows (80B)
  __shared__ float wtab[64];                        // separable gaussian: w = wtab[|dx|]*wtab[|dy|]
  int t = blockIdx.y;
  int lane = threadIdx.x & 31, wav = threadIdx.x >> 5;
  bool lo = lane < 16;
  int lrow = lane & 15;
  int kb = lo ? 0 : 16;
  int n0 = blockIdx.x * 64 + wav * 16;

  if (threadIdx.x < 64) {
    float sdev = stdp[0];
    float d = (float)threadIdx.x;
    wtab[threadIdx.x] = __expf(-0.5f * d * d / (sdev * sdev));
  }
  __syncthreads();

  // A fragments of phi (rows n0..n0+15, K=c 0..63)
  const __bf16* prow = phi_h + (size_t)(n0 + lrow) * ICH;
  union { v16bf v; v8bf h[2]; } ua;
  ua.h[0] = *(const v8bf*)(prow + kb);
  ua.h[1] = *(const v8bf*)(prow + kb + 8);
  v16bf a0 = ua.v;
  ua.h[0] = *(const v8bf*)(prow + 32 + kb);
  ua.h[1] = *(const v8bf*)(prow + 32 + kb + 8);
  v16bf a1 = ua.v;

  v8f acc0 = {0.f,0.f,0.f,0.f,0.f,0.f,0.f,0.f};
  v8f acc1 = acc0, acc2 = acc0, acc3 = acc0;
  float M[8], L[8];
  int xni[8], yni[8];
#pragma unroll
  for (int v = 0; v < 8; v++) {
    M[v] = -__builtin_inff(); L[v] = 0.f;
    int r = n0 + (lo ? v : v + 8);
    xni[v] = r & 63; yni[v] = r >> 6;
  }

  for (int m0 = 0; m0 < N_SP; m0 += 32) {
    const __bf16* tr0 = thetaT + ((size_t)t * N_SP + m0 + lrow) * ICH + (lo ? 0 : 16);
    const __bf16* tr1 = tr0 + (size_t)16 * ICH;
    const __bf16* gm  = gmat + ((size_t)t * ICH + lrow) * N_SP + m0 + (lo ? 0 : 16);
    if (m0 + 32 < N_SP) {                    // software prefetch of next chunk (global_prefetch)
      __builtin_prefetch((const void*)(tr0 + 32 * ICH), 0, 0);
      __builtin_prefetch((const void*)(tr1 + 32 * ICH), 0, 0);
      __builtin_prefetch((const void*)(gm + 32), 0, 0);
      __builtin_prefetch((const void*)(gm + (size_t)16 * N_SP + 32), 0, 0);
      __builtin_prefetch((const void*)(gm + (size_t)32 * N_SP + 32), 0, 0);
      __builtin_prefetch((const void*)(gm + (size_t)48 * N_SP + 32), 0, 0);
    }
    v8f zz = {0.f,0.f,0.f,0.f,0.f,0.f,0.f,0.f};
    v8f f0 = wmma_bf16(a0, *(const v16bf*)tr0, wmma_bf16(a1, *(const v16bf*)(tr0 + 32), zz));
    v8f f1 = wmma_bf16(a0, *(const v16bf*)tr1, wmma_bf16(a1, *(const v16bf*)(tr1 + 32), zz));

    float sc[8], p0[8], p1[8];
#pragma unroll
    for (int v = 0; v < 8; v++) {
      float cm = fmaxf(f0[v], f1[v]);
      cm = fmaxf(cm, __shfl_xor(cm, 1));
      cm = fmaxf(cm, __shfl_xor(cm, 2));
      cm = fmaxf(cm, __shfl_xor(cm, 4));
      cm = fmaxf(cm, __shfl_xor(cm, 8));
      float nM = fmaxf(M[v], cm);
      float s  = __expf(M[v] - nM);
      float e0 = __expf(f0[v] - nM);
      float e1 = __expf(f1[v] - nM);
      float rsum = e0 + e1;
      rsum += __shfl_xor(rsum, 1);
      rsum += __shfl_xor(rsum, 2);
      rsum += __shfl_xor(rsum, 4);
      rsum += __shfl_xor(rsum, 8);
      L[v] = L[v] * s + rsum;               // denominator: NO gaussian weight
      M[v] = nM; sc[v] = s; p0[v] = e0; p1[v] = e1;
    }
#pragma unroll
    for (int v = 0; v < 8; v++) { acc0[v] *= sc[v]; acc1[v] *= sc[v]; acc2[v] *= sc[v]; acc3[v] *= sc[v]; }

    int xm0 = (m0 + lrow) & 63,      ym0 = (m0 + lrow) >> 6;
    int xm1 = (m0 + 16 + lrow) & 63, ym1 = (m0 + 16 + lrow) >> 6;
#pragma unroll
    for (int v = 0; v < 8; v++) {
      float w0 = wtab[__builtin_abs(xni[v] - xm0)] * wtab[__builtin_abs(yni[v] - ym0)];
      float w1 = wtab[__builtin_abs(xni[v] - xm1)] * wtab[__builtin_abs(yni[v] - ym1)];
      int r = lo ? v : v + 8;
      plds[wav][r][lrow]      = (__bf16)(p0[v] * w0);   // numerator: weighted
      plds[wav][r][lrow + 16] = (__bf16)(p1[v] * w1);
    }

    // P (16x32) -> A fragment via wave-private LDS transpose
    union { v16bf v; v8bf h[2]; } up;
    up.h[0] = *(const v8bf*)&plds[wav][lrow][kb];
    up.h[1] = *(const v8bf*)&plds[wav][lrow][kb + 8];
    v16bf pa = up.v;

    acc0 = wmma_bf16(pa, *(const v16bf*)(gm),                     acc0);
    acc1 = wmma_bf16(pa, *(const v16bf*)(gm + (size_t)16 * N_SP), acc1);
    acc2 = wmma_bf16(pa, *(const v16bf*)(gm + (size_t)32 * N_SP), acc2);
    acc3 = wmma_bf16(pa, *(const v16bf*)(gm + (size_t)48 * N_SP), acc3);
  }

#pragma unroll
  for (int v = 0; v < 8; v++) {
    float inv = 1.f / L[v];
    size_t n = (size_t)(n0 + (lo ? v : v + 8));
    corr[((size_t)t * ICH +      lrow) * N_SP + n] = acc0[v] * inv;
    corr[((size_t)t * ICH + 16 + lrow) * N_SP + n] = acc1[v] * inv;
    corr[((size_t)t * ICH + 32 + lrow) * N_SP + n] = acc2[v] * inv;
    corr[((size_t)t * ICH + 48 + lrow) * N_SP + n] = acc3[v] * inv;
  }
}

// ---------------- memory-bank branch: f1 softmax, y1, qloss partials ----------------
__global__ __launch_bounds__(256) void mb_kernel(const float* __restrict__ phi_f,
                                                 const float* __restrict__ mbm,
                                                 float* __restrict__ y1T,
                                                 float* __restrict__ qpart) {
  __shared__ float prow[64];
  __shared__ float pk[256];
  __shared__ float red[256];
  int n = blockIdx.x, tid = threadIdx.x;
  if (tid < 64) prow[tid] = phi_f[(size_t)n * ICH + tid];
  __syncthreads();
  float f = 0.f;
#pragma unroll 8
  for (int c = 0; c < 64; c++) f += prow[c] * mbm[(size_t)c * MB_K + tid];
  f *= 0.125f;                                 // ic^-0.5
  red[tid] = f; __syncthreads();
  for (int s = 128; s > 0; s >>= 1) { if (tid < s) red[tid] = fmaxf(red[tid], red[tid + s]); __syncthreads(); }
  float mx = red[0]; __syncthreads();
  float e = __expf(f - mx);
  red[tid] = e; __syncthreads();
  for (int s = 128; s > 0; s >>= 1) { if (tid < s) red[tid] += red[tid + s]; __syncthreads(); }
  float inv = 1.f / red[0]; __syncthreads();
  pk[tid] = e * inv; __syncthreads();
  float q = 0.f;
  if (tid < 64) {
    float acc = 0.f;
    const float* mr = mbm + (size_t)tid * MB_K;
#pragma unroll 8
    for (int k = 0; k < MB_K; k++) acc += pk[k] * mr[k];
    y1T[(size_t)tid * N_SP + n] = acc;
    q = fabsf(prow[tid] - acc);
  }
  red[tid] = q; __syncthreads();
  for (int s = 128; s > 0; s >>= 1) { if (tid < s) red[tid] += red[tid + s]; __syncthreads(); }
  if (tid == 0) qpart[n] = red[0];
}

__global__ __launch_bounds__(256) void qreduce_kernel(const float* __restrict__ qpart,
                                                      float* __restrict__ qout) {
  __shared__ float red[256];
  int tid = threadIdx.x;
  float s = 0.f;
  for (int i = tid; i < N_SP; i += 256) s += qpart[i];
  red[tid] = s; __syncthreads();
  for (int st = 128; st > 0; st >>= 1) { if (tid < st) red[tid] += red[tid + st]; __syncthreads(); }
  if (tid == 0) qout[0] = red[0] * (1.f / (4096.f * 64.f));
}

// ---------------- final fused conv: z = Wz*corr + q + Wz1*y1 ----------------
__global__ __launch_bounds__(256) void final_kernel(const float* __restrict__ corr,
                                                    const float* __restrict__ y1T,
                                                    const float* __restrict__ x,
                                                    const float* __restrict__ wz_w,  const float* __restrict__ wz_b,
                                                    const float* __restrict__ wz1_w, const float* __restrict__ wz1_b,
                                                    float* __restrict__ zout) {
  __shared__ float scr[448][16];
  __shared__ float sy[64][16];
  int n0 = blockIdx.x * 16;
  for (int i = threadIdx.x; i < 448 * 16; i += 256) { int c = i >> 4, nl = i & 15; scr[c][nl] = corr[(size_t)c * N_SP + n0 + nl]; }
  for (int i = threadIdx.x; i < 64 * 16;  i += 256) { int c = i >> 4, nl = i & 15; sy[c][nl]  = y1T[(size_t)c * N_SP + n0 + nl]; }
  __syncthreads();
  int nl = threadIdx.x & 15;
  int og = threadIdx.x >> 4;  // 16 groups of 8 output channels
#pragma unroll
  for (int oo = 0; oo < 8; oo++) {
    int o = og * 8 + oo;
    float acc = wz_b[o] + wz1_b[o] + x[((size_t)(3 * C_CH) + o) * N_SP + n0 + nl];
    const float* wr = wz_w + (size_t)o * 448;
    for (int c = 0; c < 448; c++) acc += wr[c] * scr[c][nl];
    const float* w1r = wz1_w + (size_t)o * 64;
#pragma unroll 8
    for (int c = 0; c < 64; c++) acc += w1r[c] * sy[c][nl];
    zout[(size_t)o * N_SP + n0 + nl] = acc;
  }
}

extern "C" void kernel_launch(void* const* d_in, const int* in_sizes, int n_in,
                              void* d_out, int out_size, void* d_ws, size_t ws_size,
                              hipStream_t stream) {
  (void)in_sizes; (void)n_in; (void)out_size; (void)ws_size;
  const float* x        = (const float*)d_in[0];
  const float* gn_gamma = (const float*)d_in[1];
  const float* gn_beta  = (const float*)d_in[2];
  const float* g_w      = (const float*)d_in[3];
  const float* g_b      = (const float*)d_in[4];
  const float* theta_w  = (const float*)d_in[5];
  const float* theta_b  = (const float*)d_in[6];
  const float* phi_w    = (const float*)d_in[7];
  const float* phi_b    = (const float*)d_in[8];
  const float* wz_w     = (const float*)d_in[9];
  const float* wz_b     = (const float*)d_in[10];
  const float* wz1_w    = (const float*)d_in[11];
  const float* wz1_b    = (const float*)d_in[12];
  const float* mbm      = (const float*)d_in[13];
  const float* stdp     = (const float*)d_in[14];
  float* zout = (float*)d_out;
  float* qout = zout + (size_t)C_CH * N_SP;

  char* wsp = (char*)d_ws;
  size_t off = 0;
  auto alloc = [&](size_t b) -> char* { char* p = wsp + off; off = (off + b + 255) & ~(size_t)255; return p; };
  __bf16* hnT    = (__bf16*)alloc((size_t)T_FRAMES * N_SP * C_CH * 2);
  __bf16* thetaT = (__bf16*)alloc((size_t)T_FRAMES * N_SP * ICH * 2);
  __bf16* gmat   = (__bf16*)alloc((size_t)T_FRAMES * ICH * N_SP * 2);
  float*  phi_f  = (float*)alloc((size_t)N_SP * ICH * 4);
  __bf16* phi_h  = (__bf16*)alloc((size_t)N_SP * ICH * 2);
  float*  corr   = (float*)alloc((size_t)T_FRAMES * ICH * N_SP * 4);
  float*  y1T    = (float*)alloc((size_t)ICH * N_SP * 4);
  float*  mu     = (float*)alloc(T_FRAMES * 32 * 4);
  float*  rs     = (float*)alloc(T_FRAMES * 32 * 4);
  float*  qpart  = (float*)alloc(N_SP * 4);

  gn_stats_kernel<<<dim3(7, 32), 256, 0, stream>>>(x, mu, rs);
  gn_apply_kernel<<<dim3(7, 64), 256, 0, stream>>>(x, mu, rs, gn_gamma, gn_beta, hnT);
  proj_kernel<<<dim3(256, 15), 128, 0, stream>>>(hnT, theta_w, theta_b, g_w, g_b, phi_w, phi_b,
                                                 thetaT, gmat, phi_f, phi_h);
  attn_kernel<<<dim3(64, 7), 128, 0, stream>>>(phi_h, thetaT, gmat, stdp, corr);
  mb_kernel<<<4096, 256, 0, stream>>>(phi_f, mbm, y1T, qpart);
  qreduce_kernel<<<1, 256, 0, stream>>>(qpart, qout);
  final_kernel<<<256, 256, 0, stream>>>(corr, y1T, x, wz_w, wz_b, wz1_w, wz1_b, zout);
}
